// SpikingCNN_36292473651224
// MI455X (gfx1250) — matrix operand order repaired and checked by reading, not compiled, not run on hardware
//
#include <hip/hip_runtime.h>

typedef __attribute__((ext_vector_type(16))) _Float16 v16h;
typedef __attribute__((ext_vector_type(8)))  float    v8f;

#define F1C   64
#define F2C   128
#define CIN   8
#define LEN   1024
#define KW    36
#define BATCH 32
#define TSTEP 50
#define OUTN  10
#define BETA  0.9f
#define THR   1.0f

// A-fragment K index for V_WMMA_F32_16X16X32_F16 (16-bit A 16x32 layout):
// element j: K = j + (lane>=16)*16 (contiguous, matches B fragments)
__device__ __forceinline__ int kidxA(int lane, int j) {
    return j + ((lane >> 4) << 4);
}

// ---------------------------------------------------------------------------
// Pre-swizzle w1/w2 (f32) into per-lane-contiguous f16 A-fragments.
// w1frag: [4 f1tiles][9 chunks][32 lanes][16 halves]   (chunk r = k*8 + c)
// w2frag: [8 f2tiles][72 chunks][32 lanes][16 halves]  (chunk = k*2 + f1half)
// ---------------------------------------------------------------------------
__global__ __launch_bounds__(256)
void pack_weights(const float* __restrict__ w1, const float* __restrict__ w2,
                  _Float16* __restrict__ w1f, _Float16* __restrict__ w2f) {
    int idx = blockIdx.x * 256 + threadIdx.x;
    if (idx < 4 * 9 * 32 * 16) {
        int j = idx & 15, lane = (idx >> 4) & 31;
        int g = idx >> 9;             // tile*9 + chunk
        int chunk = g % 9, tile = g / 9;
        int f1 = tile * 16 + (lane & 15);
        int r = chunk * 32 + kidxA(lane, j);
        int k = r >> 3, c = r & 7;    // r = k*8 + c
        w1f[idx] = (_Float16)w1[(f1 * CIN + c) * KW + k];
    }
    if (idx < 8 * 72 * 32 * 16) {
        int j = idx & 15, lane = (idx >> 4) & 31;
        int g = idx >> 9;             // tile*72 + chunk
        int chunk = g % 72, tile = g / 72;
        int f2 = tile * 16 + (lane & 15);
        int k = chunk >> 1;
        int f1 = ((chunk & 1) << 5) + kidxA(lane, j);
        w2f[idx] = (_Float16)w2[(f2 * F1C + f1) * KW + k];
    }
}

// ---------------------------------------------------------------------------
// conv1 (C=8,K=36,'same') + bias + LIF1 -> writes s1 (f16 spikes), updates v1.
// One block = (batch b, 64-wide L tile). 8 waves; wave w: f1tile = w>>1,
// column half = (w&1)*32, two 16x16 accumulators.
// LDS slab xs[p][c] (p = L-local 0..99 incl halo, c = channel) so a B fragment
// is 16 contiguous halves: xs[(col+n)*8 + chunk*32 + 16*(lane>=16) .. +15].
// ---------------------------------------------------------------------------
__global__ __launch_bounds__(256)
void conv1_lif(const float* __restrict__ x, const _Float16* __restrict__ w1f,
               const float* __restrict__ b1, float* __restrict__ v1,
               _Float16* __restrict__ s1, int t) {
    __shared__ _Float16 xs[100 * CIN];
    int b  = blockIdx.x >> 4;
    int l0 = (blockIdx.x & 15) * 64;
    int tid = threadIdx.x;

    const float* xb = x + ((size_t)(b * TSTEP + t) * CIN) * LEN;
    for (int i = tid; i < CIN * 100; i += 256) {
        int c = i / 100, p = i % 100;          // coalesced in p per channel row
        int l = l0 - 18 + p;
        float val = (l >= 0 && l < LEN) ? xb[c * LEN + l] : 0.0f;
        xs[p * CIN + c] = (_Float16)val;
    }
    __syncthreads();

    int wave = tid >> 5, lane = tid & 31;
    int f1t  = wave >> 1;
    int colh = (wave & 1) * 32;
    int n    = lane & 15;
    v8f acc0 = {}, acc1 = {};

    for (int chunk = 0; chunk < 9; ++chunk) {
        const v16h a = *(const v16h*)(w1f + (((size_t)f1t * 9 + chunk) * 32 + lane) * 16);
        int rbase = chunk * 32 + ((lane >> 4) << 4);
        const _Float16* p0 = xs + (colh + 0  + n) * CIN + rbase;
        const _Float16* p1 = xs + (colh + 16 + n) * CIN + rbase;
        v16h bm0, bm1;
#pragma unroll
        for (int j = 0; j < 16; ++j) { bm0[j] = p0[j]; bm1[j] = p1[j]; }
        acc0 = __builtin_amdgcn_wmma_f32_16x16x32_f16(false, a, false, bm0, (short)0, acc0, false, false);
        acc1 = __builtin_amdgcn_wmma_f32_16x16x32_f16(false, a, false, bm1, (short)0, acc1, false, false);
    }

    // LIF1: v' = beta*v + h; spk = (v' > thr); v = v' - spk*thr
    int mofs = (lane >> 4) << 3;
#pragma unroll
    for (int r = 0; r < 8; ++r) {
        int f1 = f1t * 16 + r + mofs;
        float bias = b1[f1];
#pragma unroll
        for (int ct = 0; ct < 2; ++ct) {
            int l = l0 + colh + ct * 16 + n;
            float h = (ct ? acc1[r] : acc0[r]) + bias;
            size_t vi = ((size_t)(b * F1C + f1)) * LEN + l;
            float vnew = BETA * v1[vi] + h;
            float spk = vnew > THR ? 1.0f : 0.0f;
            v1[vi] = vnew - spk * THR;
            s1[vi] = (_Float16)spk;
        }
    }
}

// ---------------------------------------------------------------------------
// conv2 (F1=64,K=36,'same') + bias + LIF2 -> updates v2, accumulates spike
// counts. Block = (b, 64-wide L tile). Wave w owns f2 rows [16w,16w+16) x 64
// cols -> 4 accumulators sharing one A fragment per chunk.
// LDS slab ss[p][f1] so each B fragment is 16 contiguous halves (32B aligned).
// ---------------------------------------------------------------------------
__global__ __launch_bounds__(256)
void conv2_lif(const _Float16* __restrict__ s1, const _Float16* __restrict__ w2f,
               const float* __restrict__ b2, float* __restrict__ v2,
               float* __restrict__ cnt) {
    __shared__ _Float16 ss[100 * F1C];   // 12.5 KB
    int b  = blockIdx.x >> 4;
    int l0 = (blockIdx.x & 15) * 64;
    int tid = threadIdx.x;

    const _Float16* sb = s1 + ((size_t)b * F1C) * LEN;
    for (int i = tid; i < F1C * 100; i += 256) {
        int f = i / 100, p = i % 100;          // coalesced in p per f1 row
        int l = l0 - 18 + p;
        ss[p * F1C + f] = (l >= 0 && l < LEN) ? sb[f * LEN + l] : (_Float16)0.0f;
    }
    __syncthreads();

    int wave = tid >> 5, lane = tid & 31;
    int n = lane & 15;
    v8f acc[4] = {{}, {}, {}, {}};

    for (int chunk = 0; chunk < 72; ++chunk) {
        const v16h a = *(const v16h*)(w2f + (((size_t)wave * 72 + chunk) * 32 + lane) * 16);
        int k = chunk >> 1;
        int f1b = ((chunk & 1) << 5) + ((lane >> 4) << 4);
#pragma unroll
        for (int ct = 0; ct < 4; ++ct) {
            const v16h bm = *(const v16h*)(ss + (ct * 16 + n + k) * F1C + f1b);
            acc[ct] = __builtin_amdgcn_wmma_f32_16x16x32_f16(false, a, false, bm, (short)0, acc[ct], false, false);
        }
    }

    int mofs = (lane >> 4) << 3;
#pragma unroll
    for (int r = 0; r < 8; ++r) {
        int f2 = wave * 16 + r + mofs;
        float bias = b2[f2];
#pragma unroll
        for (int ct = 0; ct < 4; ++ct) {
            int l = l0 + ct * 16 + n;
            size_t vi = ((size_t)(b * F2C + f2)) * LEN + l;
            float vnew = BETA * v2[vi] + acc[ct][r] + bias;
            float spk = vnew > THR ? 1.0f : 0.0f;
            v2[vi] = vnew - spk * THR;
            cnt[vi] += spk;
        }
    }
}

// ---------------------------------------------------------------------------
// out[b,o] = bf[o] + (1/T) * sum_{f2,l} cnt[b,f2,l] * wf[o, f2*L + l]
// ---------------------------------------------------------------------------
__global__ __launch_bounds__(256)
void fc_out(const float* __restrict__ cnt, const float* __restrict__ wf,
            const float* __restrict__ bf, float* __restrict__ out) {
    int b = blockIdx.x / OUTN;
    int o = blockIdx.x % OUTN;
    const float* cb = cnt + (size_t)b * F2C * LEN;
    const float* wo = wf  + (size_t)o * F2C * LEN;
    float s = 0.0f;
    for (int i = threadIdx.x; i < F2C * LEN; i += 256) s += cb[i] * wo[i];
    __shared__ float red[256];
    red[threadIdx.x] = s;
    __syncthreads();
    for (int st = 128; st > 0; st >>= 1) {
        if (threadIdx.x < st) red[threadIdx.x] += red[threadIdx.x + st];
        __syncthreads();
    }
    if (threadIdx.x == 0) out[b * OUTN + o] = red[0] * (1.0f / (float)TSTEP) + bf[o];
}

extern "C" void kernel_launch(void* const* d_in, const int* in_sizes, int n_in,
                              void* d_out, int out_size, void* d_ws, size_t ws_size,
                              hipStream_t stream) {
    (void)in_sizes; (void)n_in; (void)out_size; (void)ws_size;
    const float* x  = (const float*)d_in[0];
    const float* w1 = (const float*)d_in[1];
    const float* b1 = (const float*)d_in[2];
    const float* w2 = (const float*)d_in[3];
    const float* b2 = (const float*)d_in[4];
    const float* wf = (const float*)d_in[5];
    const float* bf = (const float*)d_in[6];
    float* out = (float*)d_out;

    char* ws = (char*)d_ws;
    size_t off = 0;
    float*    v1   = (float*)(ws + off);    off += (size_t)BATCH * F1C * LEN * 4;  // 8 MB
    float*    v2   = (float*)(ws + off);    off += (size_t)BATCH * F2C * LEN * 4;  // 16 MB
    float*    cnt  = (float*)(ws + off);    off += (size_t)BATCH * F2C * LEN * 4;  // 16 MB
    _Float16* s1   = (_Float16*)(ws + off); off += (size_t)BATCH * F1C * LEN * 2;  // 4 MB
    _Float16* w1f  = (_Float16*)(ws + off); off += (size_t)4 * 9 * 32 * 16 * 2;
    _Float16* w2f  = (_Float16*)(ws + off); off += (size_t)8 * 72 * 32 * 16 * 2;

    // LIF state + accumulator must start at zero every call (deterministic).
    hipMemsetAsync(v1,  0, (size_t)BATCH * F1C * LEN * 4, stream);
    hipMemsetAsync(v2,  0, (size_t)BATCH * F2C * LEN * 4, stream);
    hipMemsetAsync(cnt, 0, (size_t)BATCH * F2C * LEN * 4, stream);

    pack_weights<<<1152, 256, 0, stream>>>(w1, w2, w1f, w2f);

    for (int t = 0; t < TSTEP; ++t) {
        conv1_lif<<<BATCH * 16, 256, 0, stream>>>(x, w1f, b1, v1, s1, t);
        conv2_lif<<<BATCH * 16, 256, 0, stream>>>(s1, w2f, b2, v2, cnt);
    }

    fc_out<<<BATCH * OUTN, 256, 0, stream>>>(cnt, wf, bf, out);
}
